// BaseCrystalModel_45449343926352
// MI455X (gfx1250) — compile-verified, matrix operand order checked by reading, not run on hardware
//
#include <hip/hip_runtime.h>
#include <hip/hip_bf16.h>

#define NNODES 50000
#define NEDGES 1000000
#define NLAYER 3
#define NGRAPH 512
#define HDIM   128
#define NG     50
#define KATTR  64      // NG padded to WMMA K granularity
#define RO1    640     // 5*H

typedef _Float16 h16 __attribute__((ext_vector_type(16)));
typedef _Float16 h2t __attribute__((ext_vector_type(2)));
typedef float    f8  __attribute__((ext_vector_type(8)));

__device__ __forceinline__ float sspf(float x) {
    // softplus(x) - log(2) via raw v_exp_f32 / v_log_f32 (base-2, no denorm
    // guards needed: 1 + 2^(-|x|*log2e) is in (1,2], never subnormal).
    float t = __builtin_amdgcn_exp2f(-1.4426950408889634f * fabsf(x));
    return fmaxf(x, 0.0f) + 0.69314718055994531f * (__builtin_amdgcn_logf(1.0f + t) - 1.0f);
}

__device__ __forceinline__ void lds_fence_wave() {
    __builtin_amdgcn_wave_barrier();
    asm volatile("s_wait_dscnt 0" ::: "memory");
    __builtin_amdgcn_wave_barrier();
}

// Async DMA: 16 bytes global -> LDS, tracked by ASYNCcnt (GVS mode:
// mem = SADDR64 + VADDR32; LDS dest = wave LDS base + VDST VGPR value).
__device__ __forceinline__ void async_g2l_b128(unsigned lds_off, unsigned gbyte_off,
                                               const void* sbase) {
    asm volatile("global_load_async_to_lds_b128 %0, %1, %2"
                 :: "v"(lds_off), "v"(gbyte_off), "s"(sbase) : "memory");
}
__device__ __forceinline__ void wait_async0() {
    asm volatile("s_wait_asynccnt 0" ::: "memory");
}

// Load a 16x32 f16 WMMA operand fragment from a row-major LDS tile.
// ISA layout (CDNA5 ISA): lanes 0-15 hold rows M=0..15 with K in
// {0..7} U {16..23}; lanes 16-31 hold the same rows with K {8..15} U {24..31}.
// Works for both A (row=M) and B when B is staged TRANSPOSED (row=N).
__device__ __forceinline__ h16 frag_ld(const _Float16* base, int stride, int lane, int k0) {
    const _Float16* rp = base + (lane & 15) * stride + k0 + ((lane >> 4) & 1) * 8;
    h16 a;
#pragma unroll
    for (int g = 0; g < 8; ++g) {
        int k = (g >> 2) * 16 + (g & 3) * 2;
        h2t p = *(const h2t*)(rp + k);
        a[2 * g]     = p.x;
        a[2 * g + 1] = p.y;
    }
    return a;
}

__device__ __forceinline__ f8 wmma_f16(h16 a, h16 b, f8 c) {
    return __builtin_amdgcn_wmma_f32_16x16x32_f16(false, a, false, b, (short)0, c, false, false);
}

// ---------------------------------------------------------------------------
// h = relu(emb[z])
// ---------------------------------------------------------------------------
__global__ __launch_bounds__(256) void embed_kernel(const int* __restrict__ z,
                                                    const float* __restrict__ emb,
                                                    float* __restrict__ h, int n) {
    int i = blockIdx.x * 256 + threadIdx.x;
    if (i < n * HDIM) {
        int node = i >> 7, c = i & 127;
        float v = emb[z[node] * HDIM + c];
        h[i] = fmaxf(v, 0.0f);
    }
}

__global__ __launch_bounds__(256) void zero_kernel(float* __restrict__ p, int n) {
    int i = blockIdx.x * 256 + threadIdx.x;
    if (i < n) p[i] = 0.0f;
}

// ---------------------------------------------------------------------------
// Generic node GEMM:  out = act(A[rows,128] @ W[128,NCOLS] + bias)
//   ACT: 0 = none, 1 = shifted-softplus
//   RESID: out[row,col] += result
//   SCATTER: atomically add result rows into pooled[seg[row], col]
// 256 threads (8 waves), 128 rows per block, WMMA f16 16x16x32.
// 4 N-tiles per iteration with interleaved accumulator chains: 3 independent
// WMMAs (+ frag_ld VALU) fill every WMMA->WMMA RAW hazard window, and one
// A fragment feeds 4 WMMAs.
// ---------------------------------------------------------------------------
template <int NCOLS, int ACT, bool RESID, bool SCATTER>
__global__ __launch_bounds__(256) void gemm_node_kernel(
    const float* __restrict__ A, int rows,
    const float* __restrict__ W,     // [128, NCOLS] row-major
    const float* __restrict__ bias,  // [NCOLS] or nullptr
    float* __restrict__ out,         // [rows, NCOLS] (unused if SCATTER)
    float* __restrict__ pooled,      // [NGRAPH, NCOLS] if SCATTER
    const int* __restrict__ seg)     // [rows] if SCATTER
{
    __shared__ _Float16 At[128 * HDIM];   // A tile, row-major f16
    __shared__ _Float16 Bt[128 * HDIM];   // W chunk, TRANSPOSED: Bt[n][k]

    const int tid = threadIdx.x;
    const int lane = tid & 31;
    const int wv = tid >> 5;
    const int hi = (lane >> 4) & 1;
    const int rowBase = blockIdx.x * 128;

    // stage A (pad missing rows with 0)
    for (int i = tid; i < 128 * (HDIM / 2); i += 256) {
        int r = i / (HDIM / 2);
        int k2 = (i % (HDIM / 2)) * 2;
        int gr = rowBase + r;
        float v0 = 0.0f, v1 = 0.0f;
        if (gr < rows) { v0 = A[gr * HDIM + k2]; v1 = A[gr * HDIM + k2 + 1]; }
        h2t p; p.x = (_Float16)v0; p.y = (_Float16)v1;
        *(h2t*)&At[r * HDIM + k2] = p;
    }

    const int nchunks = NCOLS / 128;
    for (int ch = 0; ch < nchunks; ++ch) {
        __syncthreads();   // A staged / previous chunk consumed
        for (int i = tid; i < 128 * 128; i += 256) {
            int k = i >> 7;
            int n = i & 127;                       // coalesced over n
            Bt[n * HDIM + k] = (_Float16)W[k * NCOLS + ch * 128 + n];
        }
        __syncthreads();

        const _Float16* Aw = &At[(wv * 16) * HDIM];
#pragma unroll 1
        for (int half = 0; half < 2; ++half) {
            const int ntb = half * 4;
            const int colb = ch * 128 + ntb * 16 + (lane & 15);
            f8 cA[4];
#pragma unroll
            for (int t = 0; t < 4; ++t) {
                float bv = bias ? bias[colb + t * 16] : 0.0f;
#pragma unroll
                for (int i = 0; i < 8; ++i) cA[t][i] = bv;
            }
#pragma unroll
            for (int kk = 0; kk < 4; ++kk) {
                h16 a = frag_ld(Aw, HDIM, lane, kk * 32);
#pragma unroll
                for (int t = 0; t < 4; ++t) {
                    h16 b = frag_ld(&Bt[((ntb + t) * 16) * HDIM], HDIM, lane, kk * 32);
                    cA[t] = wmma_f16(a, b, cA[t]);
                }
            }
#pragma unroll
            for (int r = 0; r < 8; ++r) {
                int row = rowBase + wv * 16 + r + hi * 8;
                if (row < rows) {
                    int g = SCATTER ? seg[row] : 0;
#pragma unroll
                    for (int t = 0; t < 4; ++t) {
                        float v = cA[t][r];
                        if (ACT == 1) v = sspf(v);
                        int col = colb + t * 16;
                        if (SCATTER) {
                            unsafeAtomicAdd(&pooled[g * NCOLS + col], v);
                        } else if (RESID) {
                            out[row * NCOLS + col] += v;
                        } else {
                            out[row * NCOLS + col] = v;
                        }
                    }
                }
            }
        }
    }
}

// ---------------------------------------------------------------------------
// Fused SchNet edge kernel (one interaction layer's message pass):
//   Wf  = (ssp(edge_attr @ W1 + b1) @ W2 + b2) * cosine_cutoff(edge_weight)
//   agg[dst] += x[src] * Wf          (f32 hardware atomics)
// 256 threads = 8 waves; each wave owns 16 edges; W1 staged per block via
// GLOBAL_LOAD_ASYNC_TO_LDS (ASYNCcnt), then transposed/converted in LDS.
// ---------------------------------------------------------------------------
#define OFF_W1F  0                              // f32 [50][128]  = 25600 B
#define OFF_W1T  (OFF_W1F + NG * HDIM * 4)      // f16 [128][64]  = 16384 B
#define OFF_W2T  (OFF_W1T + HDIM * KATTR * 2)   // f16 [128][128] = 32768 B
#define OFF_ATTR (OFF_W2T + HDIM * HDIM * 2)    // 8 x f16 [16][64]  = 16384 B
#define OFF_MID  (OFF_ATTR + 8 * 16 * KATTR * 2)// 8 x f16 [16][128] = 32768 B
#define SMEM_SZ  (OFF_MID + 8 * 16 * HDIM * 2)  // 123904 B total

__global__ __launch_bounds__(256) void edge_kernel(
    const float* __restrict__ edge_attr,    // [E, NG]
    const float* __restrict__ edge_weight,  // [E]
    const int* __restrict__ edge_src,
    const int* __restrict__ edge_dst,
    const float* __restrict__ w1,           // [NG, 128]  (layer slice)
    const float* __restrict__ b1,           // [128]
    const float* __restrict__ w2,           // [128, 128]
    const float* __restrict__ b2,           // [128]
    const float* __restrict__ xin,          // [N, 128]  (h @ cf_w1)
    float* __restrict__ agg,                // [N, 128]
    int nedges)
{
    __shared__ char smem[SMEM_SZ];          // single LDS object => offset 0

    float*    W1f = (float*)(smem + OFF_W1F);
    _Float16* W1t = (_Float16*)(smem + OFF_W1T);
    _Float16* W2t = (_Float16*)(smem + OFF_W2T);

    const int tid = threadIdx.x;
    const int lane = tid & 31;
    const int wv = tid >> 5;
    const int hi = (lane >> 4) & 1;
    const int ew = blockIdx.x * 128 + wv * 16;  // this wave's first edge

    _Float16* atw = (_Float16*)(smem + OFF_ATTR) + wv * 16 * KATTR;
    _Float16* mw  = (_Float16*)(smem + OFF_MID)  + wv * 16 * HDIM;

    // async DMA: raw f32 W1 (contiguous 25600 B) -> LDS, 16 B per op
    for (int i = tid * 4; i < NG * HDIM; i += 1024) {
        async_g2l_b128((unsigned)(OFF_W1F + i * 4), (unsigned)(i * 4), w1);
    }
    // stage W2 transposed f16 (direct path)
    for (int i = tid; i < 128 * 128; i += 256) {
        int k = i >> 7;
        int n = i & 127;
        W2t[n * HDIM + k] = (_Float16)w2[k * 128 + n];
    }
    // stage this wave's 16 edges of edge_attr (zero-pad K and tail edges)
    for (int i = lane; i < 16 * KATTR; i += 32) {
        int r = i >> 6;
        int k = i & 63;
        int e = ew + r;
        float v = (e < nedges && k < NG) ? edge_attr[e * NG + k] : 0.0f;
        atw[r * KATTR + k] = (_Float16)v;
    }
    wait_async0();            // own wave's DMA complete
    __syncthreads();          // => all of W1f resident

    // transpose/convert W1f -> W1t (f16, zero-pad K 50..63)
    for (int i = tid; i < 128 * KATTR; i += 256) {
        int k = i >> 7;
        int n = i & 127;
        float v = (k < NG) ? W1f[k * 128 + n] : 0.0f;
        W1t[n * KATTR + k] = (_Float16)v;
    }
    __syncthreads();

    // per-lane epilogue metadata for the 8 edge-rows this lane touches
    int srcs[8], dsts[8];
    float cc[8];
#pragma unroll
    for (int r = 0; r < 8; ++r) {
        int e = ew + r + hi * 8;
        bool ok = e < nedges;
        srcs[r] = ok ? edge_src[e] : 0;
        dsts[r] = ok ? edge_dst[e] : 0;
        float wgt = ok ? edge_weight[e] : 0.0f;
        cc[r] = 0.5f * (__cosf(wgt * 0.31415926535897932f) + 1.0f);  // pi/cutoff
    }

    // GEMM1: mid = ssp(attr @ W1 + b1)   [16 x 128], K = 64
#pragma unroll 1
    for (int half = 0; half < 2; ++half) {
        const int ntb = half * 4;
        const int colb = ntb * 16 + (lane & 15);
        f8 cA[4];
#pragma unroll
        for (int t = 0; t < 4; ++t) {
            float bv = b1[colb + t * 16];
#pragma unroll
            for (int i = 0; i < 8; ++i) cA[t][i] = bv;
        }
#pragma unroll
        for (int kk = 0; kk < 2; ++kk) {
            h16 a = frag_ld(atw, KATTR, lane, kk * 32);
#pragma unroll
            for (int t = 0; t < 4; ++t) {
                h16 b = frag_ld(&W1t[((ntb + t) * 16) * KATTR], KATTR, lane, kk * 32);
                cA[t] = wmma_f16(a, b, cA[t]);
            }
        }
#pragma unroll
        for (int r = 0; r < 8; ++r) {
            int m = r + hi * 8;
#pragma unroll
            for (int t = 0; t < 4; ++t) {
                mw[m * HDIM + colb + t * 16] = (_Float16)sspf(cA[t][r]);
            }
        }
    }
    lds_fence_wave();   // wave-local RAW on mid through LDS

    // GEMM2 + cutoff + gather-modulate + scatter-add, K = 128
#pragma unroll 1
    for (int half = 0; half < 2; ++half) {
        const int ntb = half * 4;
        const int colb = ntb * 16 + (lane & 15);
        f8 cA[4];
#pragma unroll
        for (int t = 0; t < 4; ++t) {
            float bv = b2[colb + t * 16];
#pragma unroll
            for (int i = 0; i < 8; ++i) cA[t][i] = bv;
        }
#pragma unroll
        for (int kk = 0; kk < 4; ++kk) {
            h16 a = frag_ld(mw, HDIM, lane, kk * 32);
#pragma unroll
            for (int t = 0; t < 4; ++t) {
                h16 b = frag_ld(&W2t[((ntb + t) * 16) * HDIM], HDIM, lane, kk * 32);
                cA[t] = wmma_f16(a, b, cA[t]);
            }
        }
#pragma unroll
        for (int r = 0; r < 8; ++r) {
            int e = ew + r + hi * 8;
            if (e < nedges) {
                const float* xr = &xin[srcs[r] * HDIM];
                float* ar = &agg[dsts[r] * HDIM];
#pragma unroll
                for (int t = 0; t < 4; ++t) {
                    int col = colb + t * 16;
                    float v = cA[t][r] * cc[r] * xr[col];
                    unsafeAtomicAdd(&ar[col], v);
                }
            }
        }
    }
}

// ---------------------------------------------------------------------------
// out[g] = ssp(pooled[g] @ ro_w2 + ro_b2) @ ro_w3 + ro_b3
// one 128-thread block per graph; K = 640 dot per thread, then block reduce.
// ---------------------------------------------------------------------------
__global__ __launch_bounds__(128) void readout2_kernel(
    const float* __restrict__ pooled,  // [G, 640]
    const float* __restrict__ w2,      // [640, 128]
    const float* __restrict__ b2,      // [128]
    const float* __restrict__ w3,      // [128]
    const float* __restrict__ b3,      // [1]
    float* __restrict__ out)           // [G]
{
    __shared__ float red[128];
    int g = blockIdx.x;
    int j = threadIdx.x;
    const float* pr = pooled + (size_t)g * RO1;
    float acc = b2[j];
    for (int k = 0; k < RO1; ++k) acc = fmaf(pr[k], w2[k * HDIM + j], acc);
    red[j] = sspf(acc) * w3[j];
    __syncthreads();
    for (int s = 64; s > 0; s >>= 1) {
        if (j < s) red[j] += red[j + s];
        __syncthreads();
    }
    if (j == 0) out[g] = red[0] + b3[0];
}

// ---------------------------------------------------------------------------
extern "C" void kernel_launch(void* const* d_in, const int* in_sizes, int n_in,
                              void* d_out, int out_size, void* d_ws, size_t ws_size,
                              hipStream_t stream) {
    const int*   z         = (const int*)d_in[0];
    const int*   edge_src  = (const int*)d_in[1];
    const int*   edge_dst  = (const int*)d_in[2];
    const int*   batch     = (const int*)d_in[3];
    /* d_in[4] = num_graphs scalar (fixed at NGRAPH) */
    const float* edge_wgt  = (const float*)d_in[5];
    const float* edge_attr = (const float*)d_in[6];
    const float* emb       = (const float*)d_in[7];
    const float* mlp_w1    = (const float*)d_in[8];
    const float* mlp_b1    = (const float*)d_in[9];
    const float* mlp_w2    = (const float*)d_in[10];
    const float* mlp_b2    = (const float*)d_in[11];
    const float* cf_w1     = (const float*)d_in[12];
    const float* cf_w2     = (const float*)d_in[13];
    const float* cf_b2     = (const float*)d_in[14];
    const float* lin_w     = (const float*)d_in[15];
    const float* lin_b     = (const float*)d_in[16];
    const float* ro_w1     = (const float*)d_in[17];
    const float* ro_b1     = (const float*)d_in[18];
    const float* ro_w2     = (const float*)d_in[19];
    const float* ro_b2     = (const float*)d_in[20];
    const float* ro_w3     = (const float*)d_in[21];
    const float* ro_b3     = (const float*)d_in[22];

    const size_t nh = (size_t)NNODES * HDIM;
    float* h      = (float*)d_ws;
    float* xb     = h + nh;          // x buffer, reused as tmp
    float* agg    = xb + nh;
    float* pooled = agg + nh;        // [G, 640]

    const int nodeGemmBlocks = (NNODES + 127) / 128;   // 391
    const int edgeBlocks     = (NEDGES + 127) / 128;   // 7813

    // h = relu(emb[z])
    embed_kernel<<<(NNODES * HDIM + 255) / 256, 256, 0, stream>>>(z, emb, h, NNODES);

    for (int l = 0; l < NLAYER; ++l) {
        const float* cw1 = cf_w1 + (size_t)l * HDIM * HDIM;
        const float* cw2 = cf_w2 + (size_t)l * HDIM * HDIM;
        const float* cb2 = cf_b2 + (size_t)l * HDIM;
        const float* lw  = lin_w + (size_t)l * HDIM * HDIM;
        const float* lb  = lin_b + (size_t)l * HDIM;
        const float* m1  = mlp_w1 + (size_t)l * NG * HDIM;
        const float* mb1 = mlp_b1 + (size_t)l * HDIM;
        const float* m2  = mlp_w2 + (size_t)l * HDIM * HDIM;
        const float* mb2 = mlp_b2 + (size_t)l * HDIM;

        // x = h @ cf_w1 (no bias)
        gemm_node_kernel<HDIM, 0, false, false><<<nodeGemmBlocks, 256, 0, stream>>>(
            h, NNODES, cw1, nullptr, xb, nullptr, nullptr);

        zero_kernel<<<(int)((nh + 255) / 256), 256, 0, stream>>>(agg, (int)nh);

        // agg[dst] += x[src] * Wf(edge_attr, edge_weight)
        edge_kernel<<<edgeBlocks, 256, 0, stream>>>(
            edge_attr, edge_wgt, edge_src, edge_dst, m1, mb1, m2, mb2, xb, agg, NEDGES);

        // tmp = ssp(agg @ cf_w2 + cf_b2)   (tmp aliases x buffer)
        gemm_node_kernel<HDIM, 1, false, false><<<nodeGemmBlocks, 256, 0, stream>>>(
            agg, NNODES, cw2, cb2, xb, nullptr, nullptr);

        // h += tmp @ lin_w + lin_b
        gemm_node_kernel<HDIM, 0, true, false><<<nodeGemmBlocks, 256, 0, stream>>>(
            xb, NNODES, lw, lb, h, nullptr, nullptr);
    }

    // readout: pooled[batch[n]] += ssp(h @ ro_w1 + ro_b1)
    zero_kernel<<<(NGRAPH * RO1 + 255) / 256, 256, 0, stream>>>(pooled, NGRAPH * RO1);
    gemm_node_kernel<RO1, 1, false, true><<<nodeGemmBlocks, 256, 0, stream>>>(
        h, NNODES, ro_w1, ro_b1, nullptr, pooled, batch);

    // out = ssp(pooled @ ro_w2 + ro_b2) @ ro_w3 + ro_b3
    readout2_kernel<<<NGRAPH, 128, 0, stream>>>(pooled, ro_w2, ro_b2, ro_w3, ro_b3,
                                                (float*)d_out);
}
